// AdaptiveDiTBlock_35622458753334
// MI455X (gfx1250) — matrix operand order repaired and checked by reading, not compiled, not run on hardware
//
#include <hip/hip_runtime.h>
#include <hip/hip_bf16.h>
#include <math.h>

#define SQ   4096
#define HD   1536
#define NHEADS 16
#define DHEAD  96
#define H3   4608
#define H4   6144
#define H6   9216

typedef __attribute__((ext_vector_type(16))) __bf16 v16bf;
typedef __attribute__((ext_vector_type(8)))  float  v8f;

using u16 = unsigned short;
using u32 = unsigned int;

union BF16x16 { v16bf v; u16 s[16]; uint4 q[2]; };

__device__ __forceinline__ u16 f2bf(float f) {
  u32 u = __builtin_bit_cast(u32, f);
  u32 r = (u + 0x7FFFu + ((u >> 16) & 1u)) >> 16;
  return (u16)r;
}

__device__ __forceinline__ v8f wmma_bf16(v16bf a, v16bf b, v8f c) {
  return __builtin_amdgcn_wmma_f32_16x16x32_bf16(false, a, false, b, (short)0, c, false, false);
}

// low 32 bits of a flat shared-memory pointer == wave-relative LDS byte offset
__device__ __forceinline__ unsigned lds_lo(const void* p) {
  return (unsigned)(unsigned long long)p;
}

// CDNA5 async copy: LDS[vdst] = MEM[saddr + voff] (16B per lane, ASYNCcnt-tracked)
__device__ __forceinline__ void async_ld_b128(unsigned lds_addr, unsigned goff,
                                              const void* base) {
  asm volatile("global_load_async_to_lds_b128 %0, %1, %2"
               :: "v"(lds_addr), "v"(goff), "s"(base) : "memory");
}

__device__ __forceinline__ void wait_asynccnt0() {
  asm volatile("s_wait_asynccnt 0x0" ::: "memory");
}

// ---------------- tiny kernels ----------------

__global__ void cond_kernel(const float* __restrict__ t_emb,
                            const int* __restrict__ labels,
                            const float* __restrict__ class_emb,
                            float* __restrict__ cond) {
  int i = blockIdx.x * 256 + threadIdx.x;
  if (i < HD) {
    float x = t_emb[i] + class_emb[labels[0] * HD + i];
    cond[i] = x / (1.f + __expf(-x));   // silu
  }
}

__global__ void mods_kernel(const float* __restrict__ cond,
                            const float* __restrict__ Wc,
                            const float* __restrict__ bc,
                            float* __restrict__ mods) {
  int j = blockIdx.x * 256 + threadIdx.x;
  if (j < H6) {
    float acc = bc[j];
    for (int i = 0; i < HD; ++i) acc += cond[i] * Wc[i * H6 + j];
    mods[j] = acc;
  }
}

__global__ void cvt_bf16_kernel(const float* __restrict__ src,
                                u16* __restrict__ dst, int n) {
  int i = blockIdx.x * 256 + threadIdx.x;
  if (i < n) dst[i] = f2bf(src[i]);
}

// ---------------- LayerNorm + adaLN modulation -> bf16 ----------------

__global__ __launch_bounds__(256)
void ln_mod_kernel(const float* __restrict__ x,
                   const float* __restrict__ mods,
                   int shOff, int scOff,
                   u16* __restrict__ out) {
  int s = blockIdx.x;
  int t = threadIdx.x;
  float vals[6];
  float sum = 0.f, sq = 0.f;
  for (int i = 0; i < 6; ++i) {
    int j = t + i * 256;
    float v = x[(size_t)s * HD + j];
    vals[i] = v; sum += v; sq += v * v;
  }
  for (int off = 16; off >= 1; off >>= 1) {
    sum += __shfl_xor(sum, off, 32);
    sq  += __shfl_xor(sq,  off, 32);
  }
  __shared__ float red[16];
  int wv = t >> 5, ln = t & 31;
  if (ln == 0) { red[wv] = sum; red[8 + wv] = sq; }
  __syncthreads();
  sum = 0.f; sq = 0.f;
  for (int w = 0; w < 8; ++w) { sum += red[w]; sq += red[8 + w]; }
  float mu = sum * (1.f / HD);
  float var = sq * (1.f / HD) - mu * mu;
  float rs = rsqrtf(var + 1e-6f);
  for (int i = 0; i < 6; ++i) {
    int j = t + i * 256;
    float h = (vals[i] - mu) * rs;
    h = h * (1.f + mods[scOff + j]) + mods[shOff + j];
    out[(size_t)s * HD + j] = f2bf(h);
  }
}

// ---------------- WMMA GEMM, async-LDS double-buffered ----------------
// A: bf16 [M,K] row-major, B: bf16 [K,N] row-major.
// Block tile 128x128, BK=32, 8 waves (4x2), each wave 32x64 = 2x4 WMMA tiles.
// EPI: 0=QKV split, 1=Wo (resid+g1 gate), 2=W1 (gelu->bf16), 3=W2 (final resid).

template<int EPI>
__global__ __launch_bounds__(256)
void gemm_bf16_kernel(const u16* __restrict__ A, const u16* __restrict__ B,
                      int K, int N,
                      const float* __restrict__ bias,
                      const float* __restrict__ mods,
                      const float* __restrict__ resid,
                      float* __restrict__ outF,
                      u16* __restrict__ outQ, u16* __restrict__ outKT,
                      u16* __restrict__ outV, u16* __restrict__ outB) {
  __shared__ __align__(16) u16 At[2][128 * 32];
  __shared__ __align__(16) u16 Bt[2][32 * 128];

  const int m0 = blockIdx.y * 128;
  const int n0 = blockIdx.x * 128;
  const int t = threadIdx.x;
  const int wave = t >> 5, lane = t & 31;
  const int waveM = wave >> 1, waveN = wave & 1;
  const int lm = lane & 15, hi = lane >> 4;

  const unsigned atBase = lds_lo(&At[0][0]);
  const unsigned btBase = lds_lo(&Bt[0][0]);

  // issue async tile loads: A 128x32 (2x16B/thread), B 32x128 (2x16B/thread)
  auto issue_tile = [&](int buf, int k0) {
    int idx = t * 8;
#pragma unroll
    for (int pass = 0; pass < 2; ++pass) {
      int ii = idx + pass * 2048;
      int ra = ii >> 5, ca = ii & 31;
      async_ld_b128(atBase + (unsigned)(buf * 8192 + ii * 2),
                    (unsigned)((m0 + ra) * K + k0 + ca) * 2u, A);
      int rb = ii >> 7, cb = ii & 127;
      async_ld_b128(btBase + (unsigned)(buf * 8192 + ii * 2),
                    (unsigned)((k0 + rb) * N + n0 + cb) * 2u, B);
    }
  };

  v8f acc[2][4];
  v8f z = {};
#pragma unroll
  for (int mi = 0; mi < 2; ++mi)
#pragma unroll
    for (int ni = 0; ni < 4; ++ni) acc[mi][ni] = z;

  const int nkt = K / 32;
  issue_tile(0, 0);

  for (int kt = 0; kt < nkt; ++kt) {
    const int cur = kt & 1;
    wait_asynccnt0();        // this wave's tile writes to LDS complete
    __syncthreads();         // => all waves' tiles ready; prior reads of cur^1 done
    if (kt + 1 < nkt) issue_tile(cur ^ 1, (kt + 1) * 32);

    BF16x16 af[2], bfr[4];
#pragma unroll
    for (int mi = 0; mi < 2; ++mi) {
      int row = waveM * 32 + mi * 16 + lm;
      const u16* p = &At[cur][row * 32 + hi * 8];
      af[mi].q[0] = *(const uint4*)p;        // K = lo..lo+7
      af[mi].q[1] = *(const uint4*)(p + 16); // K = 16+lo..23+lo
    }
#pragma unroll
    for (int ni = 0; ni < 4; ++ni) {
      const u16* p = &Bt[cur][waveN * 64 + ni * 16 + lm];
#pragma unroll
      for (int s = 0; s < 16; ++s) {
        int kk = (s < 8) ? (hi * 8 + s) : (16 + hi * 8 + s - 8);
        bfr[ni].s[s] = p[kk * 128];
      }
    }
#pragma unroll
    for (int mi = 0; mi < 2; ++mi)
#pragma unroll
      for (int ni = 0; ni < 4; ++ni)
        acc[mi][ni] = wmma_bf16(af[mi].v, bfr[ni].v, acc[mi][ni]);
  }

  // epilogue
#pragma unroll
  for (int mi = 0; mi < 2; ++mi) {
#pragma unroll
    for (int ni = 0; ni < 4; ++ni) {
      int col = n0 + waveN * 64 + ni * 16 + lm;
      float bval = bias[col];
#pragma unroll
      for (int r = 0; r < 8; ++r) {
        int row = m0 + waveM * 32 + mi * 16 + r + hi * 8;
        float v = acc[mi][ni][r] + bval;
        if constexpr (EPI == 0) {
          u16 b = f2bf(v);
          if (col < HD) {
            int hh = col / DHEAD, d = col % DHEAD;
            outQ[((size_t)hh * SQ + row) * DHEAD + d] = b;
          } else if (col < 2 * HD) {
            int c = col - HD; int hh = c / DHEAD, d = c % DHEAD;
            outKT[((size_t)hh * DHEAD + d) * SQ + row] = b;
          } else {
            int c = col - 2 * HD; int hh = c / DHEAD, d = c % DHEAD;
            outV[((size_t)hh * SQ + row) * DHEAD + d] = b;
          }
        } else if constexpr (EPI == 1) {
          outF[(size_t)row * HD + col] =
              resid[(size_t)row * HD + col] + mods[2 * HD + col] * v;
        } else if constexpr (EPI == 2) {
          float u = v + 0.044715f * v * v * v;
          float g = 0.5f * v * (1.f + tanhf(0.7978845608028654f * u));
          outB[(size_t)row * N + col] = f2bf(g);
        } else {
          outF[(size_t)row * HD + col] =
              resid[(size_t)row * HD + col] + mods[5 * HD + col] * v;
        }
      }
    }
  }
}

// ---------------- Flash attention (per head, 64 queries / block) ----------------

__global__ __launch_bounds__(128)
void attn_kernel(const u16* __restrict__ Qb, const u16* __restrict__ KTb,
                 const u16* __restrict__ Vb, u16* __restrict__ Ob) {
  const int h = blockIdx.y;
  const int w = threadIdx.x >> 5, lane = threadIdx.x & 31;
  const int lm = lane & 15, hi = lane >> 4;
  const int qbase = blockIdx.x * 64 + w * 16;

  const u16* Qh  = Qb  + (size_t)h * SQ * DHEAD;
  const u16* KTh = KTb + (size_t)h * DHEAD * SQ;
  const u16* Vh  = Vb  + (size_t)h * SQ * DHEAD;

  // Q fragments: 3 k-slices of 32 over DH=96
  BF16x16 qa[3];
  for (int kf = 0; kf < 3; ++kf) {
    const u16* p = Qh + (size_t)(qbase + lm) * DHEAD + kf * 32 + hi * 8;
    qa[kf].q[0] = *(const uint4*)p;
    qa[kf].q[1] = *(const uint4*)(p + 16);
  }

  v8f z = {};
  v8f o[6];
  for (int i = 0; i < 6; ++i) o[i] = z;
  float mst[8], lst[8];
  for (int r = 0; r < 8; ++r) { mst[r] = -1e30f; lst[r] = 0.f; }

  __shared__ __align__(16) u16 Plds[4][16][32];
  const float sscale = 0.10206207261596577f;  // 1/sqrt(96)

  for (int nk = 0; nk < SQ; nk += 32) {
    v8f sc0 = z, sc1 = z;
    for (int kf = 0; kf < 3; ++kf) {
      const u16* base = KTh + (size_t)(kf * 32) * SQ + nk + lm;
      BF16x16 kb0, kb1;
      for (int s = 0; s < 16; ++s) {
        int kk = (s < 8) ? (hi * 8 + s) : (16 + hi * 8 + s - 8);
        kb0.s[s] = base[(size_t)kk * SQ];
        kb1.s[s] = base[(size_t)kk * SQ + 16];
      }
      sc0 = wmma_bf16(qa[kf].v, kb0.v, sc0);
      sc1 = wmma_bf16(qa[kf].v, kb1.v, sc1);
    }
    // online softmax (row stats uniform within each 16-lane half)
    float corr[8];
    for (int r = 0; r < 8; ++r) {
      float s0 = sc0[r] * sscale, s1 = sc1[r] * sscale;
      float m = fmaxf(s0, s1);
      for (int off = 1; off < 16; off <<= 1) m = fmaxf(m, __shfl_xor(m, off, 16));
      float mn = fmaxf(mst[r], m);
      float c = __expf(mst[r] - mn);
      float p0 = __expf(s0 - mn), p1 = __expf(s1 - mn);
      sc0[r] = p0; sc1[r] = p1;
      float rs = p0 + p1;
      for (int off = 1; off < 16; off <<= 1) rs += __shfl_xor(rs, off, 16);
      lst[r] = lst[r] * c + rs;
      mst[r] = mn;
      corr[r] = c;
    }
    for (int nt2 = 0; nt2 < 6; ++nt2)
      for (int r = 0; r < 8; ++r) o[nt2][r] *= corr[r];

    // C-layout P -> LDS -> A-layout fragment (per-wave private tile, DS in-order)
    for (int r = 0; r < 8; ++r) {
      Plds[w][r + hi * 8][lm]      = f2bf(sc0[r]);
      Plds[w][r + hi * 8][16 + lm] = f2bf(sc1[r]);
    }
    BF16x16 pa;
    {
      const u16* p = &Plds[w][lm][hi * 8];
      pa.q[0] = *(const uint4*)p;
      pa.q[1] = *(const uint4*)(p + 16);
    }
    // P (16x32) x V (32x96)
    const u16* vrow = Vh + (size_t)nk * DHEAD + lm;
    for (int nt2 = 0; nt2 < 6; ++nt2) {
      BF16x16 vb;
      const u16* p = vrow + nt2 * 16;
      for (int s = 0; s < 16; ++s) {
        int kk = (s < 8) ? (hi * 8 + s) : (16 + hi * 8 + s - 8);
        vb.s[s] = p[(size_t)kk * DHEAD];
      }
      o[nt2] = wmma_bf16(pa.v, vb.v, o[nt2]);
    }
  }

  float inv[8];
  for (int r = 0; r < 8; ++r) inv[r] = 1.f / lst[r];
  for (int nt2 = 0; nt2 < 6; ++nt2)
    for (int r = 0; r < 8; ++r) {
      int row = qbase + r + hi * 8;
      Ob[(size_t)row * HD + h * DHEAD + nt2 * 16 + lm] = f2bf(o[nt2][r] * inv[r]);
    }
}

// ---------------- finalize: masks + inactive restore ----------------

__global__ __launch_bounds__(256)
void finalize_kernel(const float* __restrict__ hidden,
                     const float* __restrict__ prev,
                     const unsigned char* __restrict__ active,
                     float* __restrict__ out) {
  int s = blockIdx.x, t = threadIdx.x;
  const float* proc = out + (size_t)s * HD;
  float d = 0.f, p = 0.f;
  for (int i = t; i < HD; i += 256) {
    float pv = prev[(size_t)s * HD + i];
    float df = proc[i] - pv;
    d += df * df; p += pv * pv;
  }
  for (int off = 16; off >= 1; off >>= 1) {
    d += __shfl_xor(d, off, 32);
    p += __shfl_xor(p, off, 32);
  }
  __shared__ float red[16];
  int wv = t >> 5, ln = t & 31;
  if (ln == 0) { red[wv] = d; red[8 + wv] = p; }
  __syncthreads();
  d = 0.f; p = 0.f;
  for (int w = 0; w < 8; ++w) { d += red[w]; p += red[8 + w]; }
  float dn = sqrtf(d), pn = sqrtf(p) + 1e-6f;
  float cons = 1.f - fminf(fmaxf(dn / pn, 0.f), 1.f);
  bool act = active[s] != 0;
  bool conv = (cons > 0.95f) && act;
  if (!act)
    for (int i = t; i < HD; i += 256)
      out[(size_t)s * HD + i] = hidden[(size_t)s * HD + i];
  if (t == 0) {
    out[(size_t)SQ * HD + s]      = (act && !conv) ? 1.f : 0.f;  // new_active
    out[(size_t)SQ * HD + SQ + s] = conv ? 1.f : 0.f;            // exit
  }
}

// ---------------- host launcher ----------------

extern "C" void kernel_launch(void* const* d_in, const int* in_sizes, int n_in,
                              void* d_out, int out_size, void* d_ws, size_t ws_size,
                              hipStream_t stream) {
  const float* hidden   = (const float*)d_in[0];
  const float* t_emb    = (const float*)d_in[1];
  const int*   labels   = (const int*)d_in[2];
  const float* prev     = (const float*)d_in[3];
  const unsigned char* active = (const unsigned char*)d_in[4];
  const float* class_emb= (const float*)d_in[5];
  const float* W_cond   = (const float*)d_in[6];
  const float* b_cond   = (const float*)d_in[7];
  const float* Wqkv     = (const float*)d_in[8];
  const float* bqkv     = (const float*)d_in[9];
  const float* Wo       = (const float*)d_in[10];
  const float* bo       = (const float*)d_in[11];
  const float* W1       = (const float*)d_in[12];
  const float* b1       = (const float*)d_in[13];
  const float* W2       = (const float*)d_in[14];
  const float* b2       = (const float*)d_in[15];
  float* out = (float*)d_out;

  char* ws = (char*)d_ws;
  size_t off = 0;
  auto alloc = [&](size_t bytes) { size_t o = off; off += (bytes + 255) & ~(size_t)255; return o; };

  const size_t SH = (size_t)SQ * HD;
  float* cond  = (float*)(ws + alloc(HD * 4));
  float* mods  = (float*)(ws + alloc(H6 * 4));
  u16* h1      = (u16*)(ws + alloc(SH * 2));
  u16* WqkvB   = (u16*)(ws + alloc((size_t)HD * H3 * 2));
  u16* Qb      = (u16*)(ws + alloc(SH * 2));
  u16* KTb     = (u16*)(ws + alloc(SH * 2));
  u16* Vb      = (u16*)(ws + alloc(SH * 2));
  u16* attnO   = (u16*)(ws + alloc(SH * 2));
  u16* WoB     = (u16*)(ws + alloc((size_t)HD * HD * 2));
  float* x1    = (float*)(ws + alloc(SH * 4));
  u16* h2      = (u16*)(ws + alloc(SH * 2));
  u16* W1B     = (u16*)(ws + alloc((size_t)HD * H4 * 2));
  u16* mid     = (u16*)(ws + alloc((size_t)SQ * H4 * 2));
  u16* W2B     = (u16*)(ws + alloc((size_t)H4 * HD * 2));
  (void)ws_size; (void)n_in; (void)in_sizes; (void)out_size;

  cond_kernel<<<6, 256, 0, stream>>>(t_emb, labels, class_emb, cond);
  mods_kernel<<<36, 256, 0, stream>>>(cond, W_cond, b_cond, mods);

  auto cvt = [&](const float* s, u16* dst, int n) {
    cvt_bf16_kernel<<<(n + 255) / 256, 256, 0, stream>>>(s, dst, n);
  };
  cvt(Wqkv, WqkvB, HD * H3);
  cvt(Wo,   WoB,   HD * HD);
  cvt(W1,   W1B,   HD * H4);
  cvt(W2,   W2B,   H4 * HD);

  // LN1 + mod
  ln_mod_kernel<<<SQ, 256, 0, stream>>>(hidden, mods, 0, HD, h1);
  // QKV projection -> Q / K^T / V (bf16)
  gemm_bf16_kernel<0><<<dim3(H3 / 128, SQ / 128), 256, 0, stream>>>(
      h1, WqkvB, HD, H3, bqkv, mods, nullptr, nullptr, Qb, KTb, Vb, nullptr);
  // attention
  attn_kernel<<<dim3(SQ / 64, NHEADS), 128, 0, stream>>>(Qb, KTb, Vb, attnO);
  // output projection fused with residual + g1 gate -> x1 (f32)
  gemm_bf16_kernel<1><<<dim3(HD / 128, SQ / 128), 256, 0, stream>>>(
      attnO, WoB, HD, HD, bo, mods, hidden, x1, nullptr, nullptr, nullptr, nullptr);
  // LN2 + mod
  ln_mod_kernel<<<SQ, 256, 0, stream>>>(x1, mods, 3 * HD, 4 * HD, h2);
  // MLP up + GELU -> mid (bf16)
  gemm_bf16_kernel<2><<<dim3(H4 / 128, SQ / 128), 256, 0, stream>>>(
      h2, W1B, HD, H4, b1, mods, nullptr, nullptr, nullptr, nullptr, nullptr, mid);
  // MLP down fused with residual + g2 gate -> processed (f32, directly in d_out)
  gemm_bf16_kernel<3><<<dim3(HD / 128, SQ / 128), 256, 0, stream>>>(
      mid, W2B, H4, HD, b2, mods, x1, out, nullptr, nullptr, nullptr, nullptr);
  // masks + inactive restore
  finalize_kernel<<<SQ, 256, 0, stream>>>(hidden, prev, active, out);
}
